// MultiHeadAttention_89154931130794
// MI455X (gfx1250) — matrix operand order repaired and checked by reading, not compiled, not run on hardware
//
#include <hip/hip_runtime.h>
#include <hip/hip_bf16.h>
#include <math.h>

// ---------------- problem constants ----------------
constexpr int Bc = 8, Sc = 2048, Ec = 1024, Hc = 8, Dc = 128;
constexpr int E3 = 3 * Ec;          // 3072
constexpr int BS = Bc * Sc;         // 16384
constexpr float QSCALE = 0.08838834764831845f;  // 1/sqrt(128)
constexpr float LN_EPS = 1e-5f;

// ---------------- CDNA5 async copy to LDS (guarded) ----------------
#if defined(__has_builtin)
#if __has_builtin(__builtin_amdgcn_global_load_async_to_lds_b128) && \
    __has_builtin(__builtin_amdgcn_s_wait_asynccnt)
#define HAVE_ASYNC 1
#endif
#endif

typedef __attribute__((__vector_size__(4 * sizeof(int)))) int v4i;
typedef __attribute__((address_space(1))) v4i gv4i;   // global int4
typedef __attribute__((address_space(3))) v4i lv4i;   // LDS int4

// ---------------- WMMA fragment types (wave32, gfx1250) ----------------
typedef __attribute__((ext_vector_type(16))) __bf16 v16bf;
typedef __attribute__((ext_vector_type(8)))  __bf16 v8bf;
typedef __attribute__((ext_vector_type(8)))  float  v8f;

union Frag16 { v16bf v; v8bf h[2]; };

__device__ __forceinline__ v8bf ld8(const __bf16* p) {
    return *reinterpret_cast<const v8bf*>(p);
}

// A fragment: 16x32 bf16, row-major source, row = base + lane%16
// lane<16: K = {0..7, 16..23}; lane>=16: K = {8..15, 24..31}
__device__ __forceinline__ Frag16 load_a(const __bf16* row, int hi8) {
    Frag16 f;
    f.h[0] = ld8(row + hi8);
    f.h[1] = ld8(row + 16 + hi8);
    return f;
}
// B fragment: 32x16 bf16 where source is row-major [N,K] (B[k][n] = src[n][k])
__device__ __forceinline__ Frag16 load_b(const __bf16* rowNK, int hi16) {
    Frag16 f;
    f.h[0] = ld8(rowNK + hi16 / 2);
    f.h[1] = ld8(rowNK + 16 + hi16 / 2);
    return f;
}

__device__ __forceinline__ v8f wmma_bf16(const Frag16& a, const Frag16& b, v8f c) {
    return __builtin_amdgcn_wmma_f32_16x16x32_bf16(false, a.v, false, b.v,
                                                   (short)0, c, false, false);
}

// 16B global -> LDS copy: async DMA on CDNA5, sync fallback otherwise.
__device__ __forceinline__ void cp_b128(const __bf16* g, __bf16* l) {
#if HAVE_ASYNC
    __builtin_amdgcn_global_load_async_to_lds_b128((gv4i*)g, (lv4i*)l, 0, 0);
#else
    *reinterpret_cast<v8bf*>(l) = *reinterpret_cast<const v8bf*>(g);
#endif
}

// wait until ASYNCcnt <= n (immediate); no-op in fallback mode
__device__ __forceinline__ void wait_async_3() {
#if HAVE_ASYNC
    __builtin_amdgcn_s_wait_asynccnt(3);
#endif
}
__device__ __forceinline__ void wait_async_0() {
#if HAVE_ASYNC
    __builtin_amdgcn_s_wait_asynccnt(0);
#endif
}

// Stage one K-step tile pair (A: 128x32, B: 64x32) into LDS. 3 x b128 per thread.
__device__ __forceinline__ void gemm_stage(const __bf16* aBase, const __bf16* wBase,
                                           __bf16 (*As)[32], __bf16 (*Bs)[32], int t) {
    const int ra = t >> 1, ca = (t & 1) * 16;       // A: 2 threads/row, 32B each
    const __bf16* ga = aBase + (size_t)ra * Ec + ca;
    cp_b128(ga,     &As[ra][ca]);
    cp_b128(ga + 8, &As[ra][ca + 8]);
    const int rb = t >> 2, cb = (t & 3) * 8;        // B: 4 threads/row, 16B each
    cp_b128(wBase + (size_t)rb * Ec + cb, &Bs[rb][cb]);
}

// ---------------- f32 -> bf16 convert ----------------
__global__ void k_f32_to_bf16(const float* __restrict__ in, __bf16* __restrict__ out, int n) {
    int i = blockIdx.x * blockDim.x + threadIdx.x;
    int stride = gridDim.x * blockDim.x;
    for (; i < n; i += stride) out[i] = (__bf16)in[i];
}

// ---------------- QKV projection: [BS,E] x [E,3E] (+bias) -> Q,K,Vt -----------
// grid (3E/64, BS/128), block 256 (8 waves). Block tile 128x64, LDS double-buffered.
__global__ void k_qkv_gemm(const __bf16* __restrict__ xb, const __bf16* __restrict__ wb,
                           const float* __restrict__ bias,
                           __bf16* __restrict__ Qs, __bf16* __restrict__ Kb,
                           __bf16* __restrict__ Vt) {
    __shared__ __bf16 As[2][128][32];   // 16 KB
    __shared__ __bf16 Bs[2][64][32];    //  8 KB

    const int t = threadIdx.x;
    const int wave = t >> 5, lane = t & 31;
    const int lm = lane & 15, hi = (lane >= 16);
    const int m0 = blockIdx.y * 128;
    const int n0 = blockIdx.x * 64;

    const __bf16* aBase = xb + (size_t)m0 * Ec;
    const __bf16* wBase = wb + (size_t)n0 * Ec;

    v8f acc[4] = {};
    constexpr int NK = Ec / 32;   // 32 k-steps

    gemm_stage(aBase, wBase, As[0], Bs[0], t);
    for (int i = 0; i < NK; ++i) {
        const int cur = i & 1;
        if (i + 1 < NK) {
            gemm_stage(aBase + (i + 1) * 32, wBase + (i + 1) * 32,
                       As[cur ^ 1], Bs[cur ^ 1], t);
            wait_async_3();
        } else {
            wait_async_0();
        }
        __syncthreads();

        // hoist all fragment loads, then issue 4 back-to-back WMMAs
        const __bf16* arow = &As[cur][wave * 16 + lm][0];
        Frag16 a;
        a.h[0] = ld8(arow + (hi ? 8 : 0));
        a.h[1] = ld8(arow + 16 + (hi ? 8 : 0));
        Frag16 bf[4];
#pragma unroll
        for (int ns = 0; ns < 4; ++ns) {
            const __bf16* brow = &Bs[cur][ns * 16 + lm][0];
            bf[ns].h[0] = ld8(brow + (hi ? 8 : 0));
            bf[ns].h[1] = ld8(brow + 16 + (hi ? 8 : 0));
        }
#pragma unroll
        for (int ns = 0; ns < 4; ++ns)
            acc[ns] = wmma_bf16(a, bf[ns], acc[ns]);
        __syncthreads();
    }

    const int roff = hi ? 8 : 0;
    const int mw = m0 + wave * 16;
#pragma unroll
    for (int ns = 0; ns < 4; ++ns) {
#pragma unroll
        for (int r = 0; r < 8; ++r) {
            int gr = mw + roff + r;          // row in [0,BS)
            int gc = n0 + ns * 16 + lm;      // col in [0,3E)
            float v = acc[ns][r] + bias[gc];
            int b = gr / Sc, s = gr % Sc;
            if (gc < Ec) {
                int h = gc / Dc, d = gc % Dc;
                Qs[(((size_t)b * Hc + h) * Sc + s) * Dc + d] = (__bf16)(v * QSCALE);
            } else if (gc < 2 * Ec) {
                int c = gc - Ec, h = c / Dc, d = c % Dc;
                Kb[(((size_t)b * Hc + h) * Sc + s) * Dc + d] = (__bf16)v;
            } else {
                int c = gc - 2 * Ec, h = c / Dc, d = c % Dc;
                Vt[(((size_t)b * Hc + h) * Dc + d) * Sc + s] = (__bf16)v;
            }
        }
    }
}

// ---------------- softmax stats (online max / sumexp) ----------------
// grid = B*H*(S/16)/4 blocks, 128 threads (4 waves), wave handles one 16-row q tile.
// K tiles software-pipelined in registers.
__global__ void k_attn_stats(const __bf16* __restrict__ Qs, const __bf16* __restrict__ Kb,
                             float* __restrict__ marr, float* __restrict__ larr) {
    const int wave = threadIdx.x >> 5, lane = threadIdx.x & 31;
    const int lm = lane & 15, hi = (lane >= 16);
    const int ntq = Sc / 16;
    const int tq = blockIdx.x * 4 + wave;
    const int bh = tq / ntq, q0 = (tq % ntq) * 16;

    const __bf16* Qp = Qs + (size_t)bh * Sc * Dc;
    const __bf16* Kp = Kb + (size_t)bh * Sc * Dc;

    Frag16 a[4];
    const __bf16* qrow = Qp + (size_t)(q0 + lm) * Dc;
#pragma unroll
    for (int d = 0; d < 4; ++d) a[d] = load_a(qrow + d * 32, hi ? 8 : 0);

    v8f m, l;
#pragma unroll
    for (int r = 0; r < 8; ++r) { m[r] = -3.0e38f; l[r] = 0.0f; }

    Frag16 bc[4];
    {
        const __bf16* krow = Kp + (size_t)lm * Dc;
#pragma unroll
        for (int d = 0; d < 4; ++d) bc[d] = load_b(krow + d * 32, hi ? 16 : 0);
    }

    for (int k0 = 0; k0 < Sc; k0 += 16) {
        Frag16 bn[4];
        if (k0 + 16 < Sc) {
            const __bf16* krow = Kp + (size_t)(k0 + 16 + lm) * Dc;
            __builtin_prefetch(krow + 16 * Dc, 0, 1);
#pragma unroll
            for (int d = 0; d < 4; ++d) bn[d] = load_b(krow + d * 32, hi ? 16 : 0);
        }
        v8f sc = {};
#pragma unroll
        for (int d = 0; d < 4; ++d) sc = wmma_bf16(a[d], bc[d], sc);
#pragma unroll
        for (int r = 0; r < 8; ++r) {
            float s = sc[r];
            float mn = fmaxf(m[r], s);
            l[r] = l[r] * __expf(m[r] - mn) + __expf(s - mn);
            m[r] = mn;
        }
        if (k0 + 16 < Sc) {
#pragma unroll
            for (int d = 0; d < 4; ++d) bc[d] = bn[d];
        }
    }
    // combine across the 16 lanes of each half-wave (same 8 rows per half)
#pragma unroll
    for (int mask = 1; mask < 16; mask <<= 1) {
#pragma unroll
        for (int r = 0; r < 8; ++r) {
            float mo = __shfl_xor(m[r], mask, 32);
            float lo = __shfl_xor(l[r], mask, 32);
            float mn = fmaxf(m[r], mo);
            l[r] = l[r] * __expf(m[r] - mn) + lo * __expf(mo - mn);
            m[r] = mn;
        }
    }
    if (lm == 0) {
        int roff = hi ? 8 : 0;
#pragma unroll
        for (int r = 0; r < 8; ++r) {
            marr[(size_t)bh * Sc + q0 + roff + r] = m[r];
            larr[(size_t)bh * Sc + q0 + roff + r] = l[r];
        }
    }
}

// ---------------- probs + attn-average (LDS reduce) + ctx = P@V ----------------
// grid = B*(S/16), block 256 (8 waves, wave == head).
__global__ void k_attn_ctx(const __bf16* __restrict__ Qs, const __bf16* __restrict__ Kb,
                           const __bf16* __restrict__ Vt,
                           const float* __restrict__ marr, const float* __restrict__ larr,
                           float* __restrict__ attn_out, __bf16* __restrict__ ctxb) {
    __shared__ __bf16 pbuf[8][16][32];   // per-wave private 16x32 prob tile (bf16)
    __shared__ float attn_acc[16][32];   // head-summed probability tile

    const int t = threadIdx.x;
    const int wave = t >> 5, lane = t & 31;
    const int lm = lane & 15, hi = (lane >= 16);
    const int roff = hi ? 8 : 0;
    const int ntq = Sc / 16;
    const int b = blockIdx.x / ntq, q0 = (blockIdx.x % ntq) * 16;
    const int bh = b * Hc + wave;

    const __bf16* Qp = Qs + (size_t)bh * Sc * Dc;
    const __bf16* Kp = Kb + (size_t)bh * Sc * Dc;
    const __bf16* Vp = Vt + (size_t)bh * Dc * Sc;

    v8f mv = *reinterpret_cast<const v8f*>(marr + (size_t)bh * Sc + q0 + roff);
    v8f lv = *reinterpret_cast<const v8f*>(larr + (size_t)bh * Sc + q0 + roff);
    v8f inv;
#pragma unroll
    for (int r = 0; r < 8; ++r) inv[r] = 1.0f / lv[r];

    Frag16 a[4];
    const __bf16* qrow = Qp + (size_t)(q0 + lm) * Dc;
#pragma unroll
    for (int d = 0; d < 4; ++d) a[d] = load_a(qrow + d * 32, hi ? 8 : 0);

    v8f cacc[8] = {};
    float* attnb = attn_out + (size_t)b * Sc * Sc;

    for (int k0 = 0; k0 < Sc; k0 += 32) {
        // prefetch V fragments for this k-slab early
        Frag16 vb[8];
#pragma unroll
        for (int ns = 0; ns < 8; ++ns) {
            const __bf16* vrow = Vp + (size_t)(ns * 16 + lm) * Sc + k0;
            vb[ns] = load_b(vrow, hi ? 16 : 0);
        }

        attn_acc[t >> 4][(t & 15) * 2]     = 0.0f;
        attn_acc[t >> 4][(t & 15) * 2 + 1] = 0.0f;
        __syncthreads();

#pragma unroll
        for (int half = 0; half < 2; ++half) {
            const int kk = k0 + half * 16;
            const __bf16* krow = Kp + (size_t)(kk + lm) * Dc;
            // hoist K fragment loads, then chain the 4 WMMAs
            Frag16 bf[4];
#pragma unroll
            for (int d = 0; d < 4; ++d) bf[d] = load_b(krow + d * 32, hi ? 16 : 0);
            v8f sc = {};
#pragma unroll
            for (int d = 0; d < 4; ++d) sc = wmma_bf16(a[d], bf[d], sc);
#pragma unroll
            for (int r = 0; r < 8; ++r) {
                float p = __expf(sc[r] - mv[r]) * inv[r];
                atomicAdd(&attn_acc[roff + r][half * 16 + lm], p * (1.0f / Hc));  // ds_add_f32
                pbuf[wave][roff + r][half * 16 + lm] = (__bf16)p;
            }
        }
        __syncthreads();

        // one pass of plain stores for the head-averaged attention tile
        {
            int row = t >> 4, col = (t & 15) * 2;
            float2 v2 = make_float2(attn_acc[row][col], attn_acc[row][col + 1]);
            *reinterpret_cast<float2*>(&attnb[(size_t)(q0 + row) * Sc + k0 + col]) = v2;
        }

        // P @ V on this wave's private prob tile
        Frag16 pa;
        const __bf16* prow = &pbuf[wave][lm][0];
        pa.h[0] = ld8(prow + (hi ? 8 : 0));
        pa.h[1] = ld8(prow + 16 + (hi ? 8 : 0));
#pragma unroll
        for (int ns = 0; ns < 8; ++ns) cacc[ns] = wmma_bf16(pa, vb[ns], cacc[ns]);

        __syncthreads();
    }

#pragma unroll
    for (int ns = 0; ns < 8; ++ns) {
#pragma unroll
        for (int r = 0; r < 8; ++r) {
            int s = q0 + roff + r;
            int col = wave * Dc + ns * 16 + lm;
            ctxb[((size_t)b * Sc + s) * Ec + col] = (__bf16)cacc[ns][r];
        }
    }
}

// ---------------- out projection: [BS,E] x [E,E] (+bias) -> proj f32 ---------
// Same LDS double-buffered structure as the QKV GEMM.
__global__ void k_out_gemm(const __bf16* __restrict__ ctxb, const __bf16* __restrict__ owb,
                           const float* __restrict__ ob, float* __restrict__ proj) {
    __shared__ __bf16 As[2][128][32];
    __shared__ __bf16 Bs[2][64][32];

    const int t = threadIdx.x;
    const int wave = t >> 5, lane = t & 31;
    const int lm = lane & 15, hi = (lane >= 16);
    const int m0 = blockIdx.y * 128;
    const int n0 = blockIdx.x * 64;

    const __bf16* aBase = ctxb + (size_t)m0 * Ec;
    const __bf16* wBase = owb + (size_t)n0 * Ec;

    v8f acc[4] = {};
    constexpr int NK = Ec / 32;

    gemm_stage(aBase, wBase, As[0], Bs[0], t);
    for (int i = 0; i < NK; ++i) {
        const int cur = i & 1;
        if (i + 1 < NK) {
            gemm_stage(aBase + (i + 1) * 32, wBase + (i + 1) * 32,
                       As[cur ^ 1], Bs[cur ^ 1], t);
            wait_async_3();
        } else {
            wait_async_0();
        }
        __syncthreads();

        const __bf16* arow = &As[cur][wave * 16 + lm][0];
        Frag16 a;
        a.h[0] = ld8(arow + (hi ? 8 : 0));
        a.h[1] = ld8(arow + 16 + (hi ? 8 : 0));
        Frag16 bf[4];
#pragma unroll
        for (int ns = 0; ns < 4; ++ns) {
            const __bf16* brow = &Bs[cur][ns * 16 + lm][0];
            bf[ns].h[0] = ld8(brow + (hi ? 8 : 0));
            bf[ns].h[1] = ld8(brow + 16 + (hi ? 8 : 0));
        }
#pragma unroll
        for (int ns = 0; ns < 4; ++ns)
            acc[ns] = wmma_bf16(a, bf[ns], acc[ns]);
        __syncthreads();
    }

    const int roff = hi ? 8 : 0;
    const int mw = m0 + wave * 16;
#pragma unroll
    for (int ns = 0; ns < 4; ++ns) {
#pragma unroll
        for (int r = 0; r < 8; ++r) {
            int gr = mw + roff + r;
            int gc = n0 + ns * 16 + lm;
            proj[(size_t)gr * Ec + gc] = acc[ns][r] + ob[gc];
        }
    }
}

// ---------------- residual + LayerNorm + mean over S -> context [B,E] --------
__global__ void k_ln_mean(const float* __restrict__ x, const float* __restrict__ proj,
                          const float* __restrict__ g, const float* __restrict__ be,
                          float* __restrict__ ctx_out) {
    const int wave = threadIdx.x >> 5, lane = threadIdx.x & 31;
    const int row = blockIdx.x * 8 + wave;
    const int b = row / Sc;
    const float* xr = x + (size_t)row * Ec;
    const float* pr = proj + (size_t)row * Ec;

    float v[32];
    float sum = 0.0f;
#pragma unroll
    for (int j = 0; j < 32; ++j) {
        int c = lane + j * 32;
        float tv = xr[c] + pr[c];
        v[j] = tv; sum += tv;
    }
#pragma unroll
    for (int mask = 16; mask >= 1; mask >>= 1) sum += __shfl_xor(sum, mask, 32);
    float mu = sum * (1.0f / Ec);

    float s2 = 0.0f;
#pragma unroll
    for (int j = 0; j < 32; ++j) { float d = v[j] - mu; s2 += d * d; }
#pragma unroll
    for (int mask = 16; mask >= 1; mask >>= 1) s2 += __shfl_xor(s2, mask, 32);
    float rs = rsqrtf(s2 * (1.0f / Ec) + LN_EPS);

#pragma unroll
    for (int j = 0; j < 32; ++j) {
        int c = lane + j * 32;
        float y = (v[j] - mu) * rs * g[c] + be[c];
        atomicAdd(&ctx_out[(size_t)b * Ec + c], y * (1.0f / Sc));
    }
}

// ---------------- launcher ----------------
extern "C" void kernel_launch(void* const* d_in, const int* in_sizes, int n_in,
                              void* d_out, int out_size, void* d_ws, size_t ws_size,
                              hipStream_t stream) {
    const float* x   = (const float*)d_in[0];
    const float* ipw = (const float*)d_in[1];
    const float* ipb = (const float*)d_in[2];
    const float* ow  = (const float*)d_in[3];
    const float* ob  = (const float*)d_in[4];
    const float* lng = (const float*)d_in[5];
    const float* lnb = (const float*)d_in[6];

    auto al = [](size_t v) { return (v + 255) & ~size_t(255); };
    char* p = (char*)d_ws;
    __bf16* xb  = (__bf16*)p; p += al((size_t)BS * Ec * 2);
    __bf16* wb  = (__bf16*)p; p += al((size_t)E3 * Ec * 2);
    __bf16* owb = (__bf16*)p; p += al((size_t)Ec * Ec * 2);
    __bf16* Qs  = (__bf16*)p; p += al((size_t)BS * Ec * 2);
    __bf16* Kb  = (__bf16*)p; p += al((size_t)BS * Ec * 2);
    __bf16* Vt  = (__bf16*)p; p += al((size_t)BS * Ec * 2);
    float*  ma  = (float*)p;  p += al((size_t)Bc * Hc * Sc * 4);
    float*  la  = (float*)p;  p += al((size_t)Bc * Hc * Sc * 4);
    __bf16* cxb = (__bf16*)p; p += al((size_t)BS * Ec * 2);
    float*  prj = (float*)p;  p += al((size_t)BS * Ec * 4);

    float* ctx_out  = (float*)d_out;                 // [B, E]
    float* attn_out = (float*)d_out + Bc * Ec;       // [B, S, S], fully overwritten

    // zero only the atomic target (context vector)
    (void)hipMemsetAsync(ctx_out, 0, (size_t)Bc * Ec * sizeof(float), stream);

    // f32 -> bf16 conversions
    k_f32_to_bf16<<<4096, 256, 0, stream>>>(x, xb, BS * Ec);
    k_f32_to_bf16<<<2048, 256, 0, stream>>>(ipw, wb, E3 * Ec);
    k_f32_to_bf16<<<1024, 256, 0, stream>>>(ow, owb, Ec * Ec);

    // QKV projection + scatter (q scaled)
    k_qkv_gemm<<<dim3(E3 / 64, BS / 128), 256, 0, stream>>>(xb, wb, ipb, Qs, Kb, Vt);

    // softmax stats
    k_attn_stats<<<(Bc * Hc * (Sc / 16)) / 4, 128, 0, stream>>>(Qs, Kb, ma, la);

    // probabilities -> attn average + context
    k_attn_ctx<<<Bc * (Sc / 16), 256, 0, stream>>>(Qs, Kb, Vt, ma, la, attn_out, cxb);

    // output projection
    k_out_gemm<<<dim3(Ec / 64, BS / 128), 256, 0, stream>>>(cxb, owb, ob, prj);

    // residual + LN + mean over sequence
    k_ln_mean<<<BS / 8, 256, 0, stream>>>(x, prj, lng, lnb, ctx_out);
}
